// BandedJointEncoder_61564061221091
// MI455X (gfx1250) — matrix operand order repaired and checked by reading, not compiled, not run on hardware
//
#include <hip/hip_runtime.h>
#include <hip/hip_bf16.h>
#include <math.h>

// ---------------------------------------------------------------------------
// Problem constants (from reference): B=16, T=256, D_IN=128, H0=H1=128, Z=64
// M = B*T = 4096 flattened rows.
// ---------------------------------------------------------------------------
#define BQ   16
#define TQ   256
#define DQ   128
#define ZQ   64
#define MQ   (BQ * TQ)        // 4096

typedef __attribute__((ext_vector_type(16))) _Float16 v16h;
typedef __attribute__((ext_vector_type(8)))  _Float16 v8h;
typedef __attribute__((ext_vector_type(8)))  float    v8f;
typedef __attribute__((ext_vector_type(4)))  float    v4f;

// ---------------- WMMA tile helpers (layouts per CDNA5 ISA 7.12.2) ----------
// A (16x32 f16): lane<16 -> row=lane, elems 0..7 = K+0..7, 8..15 = K+16..23
//                lane>=16 -> row=lane-16, elems 0..7 = K+8..15, 8..15 = K+24..31
__device__ __forceinline__ v16h wmma_load_a(const _Float16* __restrict__ rowp,
                                            int khalf) {
  v8h lo = *(const v8h*)(rowp + khalf);
  v8h hi = *(const v8h*)(rowp + khalf + 16);
  v16h a;
#pragma unroll
  for (int i = 0; i < 8; ++i) { a[i] = lo[i]; a[i + 8] = hi[i]; }
  return a;
}

__device__ __forceinline__ v16h wmma_zero_a() {
  v16h a;
#pragma unroll
  for (int i = 0; i < 16; ++i) a[i] = (_Float16)0.0f;
  return a;
}

// B (32x16 f16, row-major K x N source): lane<16 -> col=lane, elems 0..7 = K+0..7, 8..15 = K+16..23
//                                        lane>=16 -> col=lane-16, elems 0..7 = K+8..15, 8..15 = K+24..31
__device__ __forceinline__ v16h wmma_load_b(const _Float16* __restrict__ W,
                                            int ldw, int kb, int kk, int col) {
  v16h b;
#pragma unroll
  for (int j = 0; j < 16; ++j) b[j] = W[(kb + (kk >> 1) + j + (j >= 8 ? 8 : 0)) * ldw + col];
  return b;
}

// ---------------- f32 -> f16 conversion ------------------------------------
__global__ void cvt_f32_to_f16_kernel(const float* __restrict__ src,
                                      _Float16* __restrict__ dst, int n) {
  int i = blockIdx.x * 256 + threadIdx.x;
  if (i < n) dst[i] = (_Float16)src[i];
}

// ---------------- conv1d (win=3, SAME) as 3-tap WMMA GEMM -------------------
// grid: (MQ/16, DQ/16), block: 32 (one wave)
__global__ void conv_wmma_kernel(const _Float16* __restrict__ Xh,  // (M, 128)
                                 const _Float16* __restrict__ CW,  // (3,128,128)
                                 const float* __restrict__ cb,     // (128)
                                 _Float16* __restrict__ H0out) {   // (M, 128)
  const int lane = threadIdx.x & 31;
  const int m0 = blockIdx.x * 16;
  const int n0 = blockIdx.y * 16;
  const int r15 = lane & 15;
  const int hi  = lane >> 4;
  const int m   = m0 + r15;
  const int bb  = m >> 8;
  const int t   = m & 255;
  const int col = n0 + r15;
  const int kk  = hi * 16;

  v8f acc = {};
#pragma unroll
  for (int w = 0; w < 3; ++w) {
    const int ts = t + w - 1;
    const bool valid = (ts >= 0) && (ts < TQ);
    const _Float16* rowp = Xh + (size_t)(bb * TQ + (valid ? ts : 0)) * DQ;
    const _Float16* Wp   = CW + (size_t)w * DQ * DQ;
#pragma unroll
    for (int kb = 0; kb < DQ; kb += 32) {
      v16h a = valid ? wmma_load_a(rowp, kb + hi * 8) : wmma_zero_a();
      v16h bm = wmma_load_b(Wp, DQ, kb, kk, col);
      acc = __builtin_amdgcn_wmma_f32_16x16x32_f16(
          false, a, false, bm, (short)0, acc, false, false);
    }
  }

  const float bv = cb[col];
#pragma unroll
  for (int r = 0; r < 8; ++r) {
    const int mr = m0 + hi * 8 + r;
    float v = fmaxf(acc[r] + bv, 0.0f);
    H0out[(size_t)mr * DQ + col] = (_Float16)v;
  }
}

// ---------------- generic K=128 WMMA GEMM (bias, optional relu) -------------
// grid: (M/16, N/16), block: 32
__global__ void wmma_gemm_kernel(const _Float16* __restrict__ A,   // (M,128)
                                 const _Float16* __restrict__ W,   // (128,N)
                                 const float* __restrict__ bias,   // (N)
                                 _Float16* __restrict__ out16,     // (M,N) or null
                                 float* __restrict__ out32,        // (M,N) or null
                                 int N, int do_relu) {
  const int lane = threadIdx.x & 31;
  const int m0 = blockIdx.x * 16;
  const int n0 = blockIdx.y * 16;
  const int r15 = lane & 15;
  const int hi  = lane >> 4;
  const int row = m0 + r15;
  const int col = n0 + r15;
  const int kk  = hi * 16;

  const _Float16* rowp = A + (size_t)row * DQ;
  v8f acc = {};
#pragma unroll
  for (int kb = 0; kb < DQ; kb += 32) {
    v16h a  = wmma_load_a(rowp, kb + hi * 8);
    v16h bm = wmma_load_b(W, N, kb, kk, col);
    acc = __builtin_amdgcn_wmma_f32_16x16x32_f16(
        false, a, false, bm, (short)0, acc, false, false);
  }

  const float bv = bias[col];
#pragma unroll
  for (int r = 0; r < 8; ++r) {
    const int mr = m0 + hi * 8 + r;
    float v = acc[r] + bv;
    if (do_relu) v = fmaxf(v, 0.0f);
    if (out16) out16[(size_t)mr * N + col] = (_Float16)v;
    else       out32[(size_t)mr * N + col] = v;
  }
}

// ---------------- mu head: WMMA GEMM storing mean transposed ----------------
// mean[b, z, t] = (h1 @ mu_w + mu_b)[b*T+t, z].  grid: (MQ/16, ZQ/16)
__global__ void mu_head_kernel(const _Float16* __restrict__ H1,  // (M,128)
                               const _Float16* __restrict__ MW,  // (128,64)
                               const float* __restrict__ mb,     // (64)
                               float* __restrict__ mean) {       // (B,Z,T)
  const int lane = threadIdx.x & 31;
  const int m0 = blockIdx.x * 16;
  const int n0 = blockIdx.y * 16;
  const int r15 = lane & 15;
  const int hi  = lane >> 4;
  const int row = m0 + r15;
  const int z   = n0 + r15;
  const int kk  = hi * 16;

  const _Float16* rowp = H1 + (size_t)row * DQ;
  v8f acc = {};
#pragma unroll
  for (int kb = 0; kb < DQ; kb += 32) {
    v16h a  = wmma_load_a(rowp, kb + hi * 8);
    v16h bm = wmma_load_b(MW, ZQ, kb, kk, z);
    acc = __builtin_amdgcn_wmma_f32_16x16x32_f16(
        false, a, false, bm, (short)0, acc, false, false);
  }

  const float bv = mb[z];
#pragma unroll
  for (int r = 0; r < 8; ++r) {
    const int mr = m0 + hi * 8 + r;
    const int bb = mr >> 8;
    const int t  = mr & 255;
    mean[((size_t)bb * ZQ + z) * TQ + t] = acc[r] + bv;
  }
}

// ---------------- scale_tril: closed-form bidiagonal inverse ----------------
// For upper-bidiagonal U (diag d, superdiag u):
//   scale_tril[r, c] = (r>=c) ? (-1)^(r-c) * (1/d_r) * exp2(S_r - S_c) : 0
// where S_t = sum_{k<t} log2(u_k / d_k).
// grid: 1024 blocks (one per (b,z)), 256 threads.
__global__ void scale_tril_kernel(const float* __restrict__ lv,   // (M, 128)
                                  float* __restrict__ out) {      // (B,Z,T,T)
  __shared__ float s_excl[TQ];
  __shared__ float s_invd[TQ];
  __shared__ float s_scan[TQ];

  const int t  = threadIdx.x;
  const int bz = blockIdx.x;
  const int bb = bz >> 6;
  const int z  = bz & 63;

  const size_t rowoff = (size_t)(bb * TQ + t) * (2 * ZQ);
  const float lvd = lv[rowoff + 2 * z];
  const float lvu = lv[rowoff + 2 * z + 1];
  // stable softplus
  const float d = fmaxf(lvd, 0.0f) + log1pf(__expf(-fabsf(lvd))) + 1.0f;
  const float u = fmaxf(lvu, 0.0f) + log1pf(__expf(-fabsf(lvu)));

  s_invd[t] = 1.0f / d;
  s_scan[t] = log2f(fmaxf(u, 1e-30f)) - log2f(d);
  __syncthreads();

  // Hillis-Steele inclusive scan over 256 elements
#pragma unroll
  for (int off = 1; off < TQ; off <<= 1) {
    const float add = (t >= off) ? s_scan[t - off] : 0.0f;
    __syncthreads();
    s_scan[t] += add;
    __syncthreads();
  }
  s_excl[t] = (t == 0) ? 0.0f : s_scan[t - 1];
  __syncthreads();

  // fill 256x256 matrix; each thread: 4 consecutive cols, 4-row interleave
  float* mat = out + (size_t)bz * TQ * TQ;
  const int c4   = (t & 63) * 4;
  const int rsub = t >> 6;
  const float Sc0 = s_excl[c4 + 0];
  const float Sc1 = s_excl[c4 + 1];
  const float Sc2 = s_excl[c4 + 2];
  const float Sc3 = s_excl[c4 + 3];

  for (int rb = 0; rb < TQ; rb += 4) {
    const int r = rb + rsub;
    const float Sr   = s_excl[r];
    const float invd = s_invd[r];
    float vals[4];
    const float sc[4] = {Sc0, Sc1, Sc2, Sc3};
#pragma unroll
    for (int j = 0; j < 4; ++j) {
      const int c = c4 + j;
      float v = 0.0f;
      if (c <= r) {
        v = invd * exp2f(Sr - sc[j]);
        if ((r ^ c) & 1) v = -v;
      }
      vals[j] = v;
    }
    v4f pack = {vals[0], vals[1], vals[2], vals[3]};
    __builtin_nontemporal_store(pack, (v4f*)(mat + (size_t)r * TQ + c4));
  }
}

// ---------------------------------------------------------------------------
extern "C" void kernel_launch(void* const* d_in, const int* in_sizes, int n_in,
                              void* d_out, int out_size, void* d_ws, size_t ws_size,
                              hipStream_t stream) {
  (void)in_sizes; (void)n_in; (void)out_size; (void)ws_size;

  const float* x      = (const float*)d_in[0];  // (16,256,128)
  const float* conv_w = (const float*)d_in[1];  // (3,128,128)
  const float* conv_b = (const float*)d_in[2];  // (128)
  const float* w1     = (const float*)d_in[3];  // (128,128)
  const float* b1     = (const float*)d_in[4];  // (128)
  const float* mu_w   = (const float*)d_in[5];  // (128,64)
  const float* mu_b   = (const float*)d_in[6];  // (64)
  const float* lv_w   = (const float*)d_in[7];  // (128,128)
  const float* lv_b   = (const float*)d_in[8];  // (128)

  // workspace layout (256B aligned offsets)
  char* ws = (char*)d_ws;
  _Float16* xh  = (_Float16*)(ws + 0);          // 524288 h = 1,048,576 B
  _Float16* cwh = (_Float16*)(ws + 1048576);    //  49152 h =    98,304 B
  _Float16* w1h = (_Float16*)(ws + 1146880);    //  16384 h =    32,768 B
  _Float16* mwh = (_Float16*)(ws + 1179648);    //   8192 h =    16,384 B
  _Float16* lwh = (_Float16*)(ws + 1196032);    //  16384 h =    32,768 B
  _Float16* h0  = (_Float16*)(ws + 1228800);    // 524288 h = 1,048,576 B
  _Float16* h1  = (_Float16*)(ws + 2277376);    // 524288 h = 1,048,576 B
  float*    lvf = (float*)   (ws + 3325952);    // 524288 f = 2,097,152 B

  // 1) conversions to f16
  cvt_f32_to_f16_kernel<<<(MQ * DQ + 255) / 256, 256, 0, stream>>>(x, xh, MQ * DQ);
  cvt_f32_to_f16_kernel<<<(3 * DQ * DQ + 255) / 256, 256, 0, stream>>>(conv_w, cwh, 3 * DQ * DQ);
  cvt_f32_to_f16_kernel<<<(DQ * DQ + 255) / 256, 256, 0, stream>>>(w1, w1h, DQ * DQ);
  cvt_f32_to_f16_kernel<<<(DQ * ZQ + 255) / 256, 256, 0, stream>>>(mu_w, mwh, DQ * ZQ);
  cvt_f32_to_f16_kernel<<<(DQ * DQ + 255) / 256, 256, 0, stream>>>(lv_w, lwh, DQ * DQ);

  // 2) conv + relu -> h0
  conv_wmma_kernel<<<dim3(MQ / 16, DQ / 16), 32, 0, stream>>>(xh, cwh, conv_b, h0);

  // 3) h1 = relu(h0 @ w1 + b1)
  wmma_gemm_kernel<<<dim3(MQ / 16, DQ / 16), 32, 0, stream>>>(
      h0, w1h, b1, h1, nullptr, DQ, 1);

  // 4) mean = transpose(h1 @ mu_w + mu_b)  -> d_out[0 : 16*64*256]
  mu_head_kernel<<<dim3(MQ / 16, ZQ / 16), 32, 0, stream>>>(
      h1, mwh, mu_b, (float*)d_out);

  // 5) logvar = h1 @ lv_w + lv_b (f32, in ws)
  wmma_gemm_kernel<<<dim3(MQ / 16, (2 * ZQ) / 16), 32, 0, stream>>>(
      h1, lwh, lv_b, nullptr, lvf, 2 * ZQ, 0);

  // 6) scale_tril closed-form fill -> d_out[16*64*256 : ]
  float* tril_out = (float*)d_out + (size_t)BQ * ZQ * TQ;
  scale_tril_kernel<<<BQ * ZQ, TQ, 0, stream>>>(lvf, tril_out);
}
